// WHTConv2D_34548716929185
// MI455X (gfx1250) — matrix operand order repaired and checked by reading, not compiled, not run on hardware
//
#include <hip/hip_runtime.h>
#include <math.h>

typedef _Float16 v16h __attribute__((ext_vector_type(16)));
typedef _Float16 v8h  __attribute__((ext_vector_type(8)));
typedef _Float16 v4h  __attribute__((ext_vector_type(4)));
typedef float    v8f  __attribute__((ext_vector_type(8)));

#define LDP 264          // padded f16 row stride (bank-conflict free: 528B -> bank stride 4)
#define IMG 65536        // 256*256

// H[i][j] = (-1)^popcount(i & j); build +-1.0h by bit trick: 0x3C00 | sign<<15
__device__ __forceinline__ _Float16 hsign(int a, int b) {
    unsigned short bits = (unsigned short)(0x3C00u | ((unsigned)(__popc(a & b) & 1) << 15));
    return __builtin_bit_cast(_Float16, bits);
}

// A-operand fragment of H: rows m0..m0+15, cols kb..kb+31 (16-bit A 16x32 layout)
__device__ __forceinline__ v16h make_hfrag_A(int lane, int m0, int kb) {
    int m    = m0 + (lane & 15);
    int half = (lane >> 4) << 3;          // 0 or 8
    v16h f;
#pragma unroll
    for (int e = 0; e < 8; ++e) {
        f[e]     = hsign(m, kb + half + e);
        f[e + 8] = hsign(m, kb + 16 + half + e);
    }
    return f;
}

// B-operand fragment of H: rows kb..kb+31, cols n0..n0+15 (row striped across lanes)
__device__ __forceinline__ v16h make_hfrag_B(int lane, int kb, int n0) {
    int n  = n0 + (lane & 15);
    int kh = kb + ((lane >> 4) << 3);     // k base 0 or +8
    v16h f;
#pragma unroll
    for (int e = 0; e < 8; ++e) { f[e] = hsign(kh + e, n); f[e + 8] = hsign(kh + 16 + e, n); }
    return f;
}

// A-operand fragment from row-major LDS buffer (stride LDP)
__device__ __forceinline__ v16h load_Afrag(const _Float16* buf, int lane, int m0, int kb) {
    int row  = m0 + (lane & 15);
    int half = (lane >> 4) << 3;
    const _Float16* p = buf + row * LDP;
    v8h lo = *(const v8h*)(p + kb + half);        // K = kb+half .. +7
    v8h hi = *(const v8h*)(p + kb + 16 + half);   // K = kb+16+half .. +7
    return __builtin_shufflevector(lo, hi, 0,1,2,3,4,5,6,7,8,9,10,11,12,13,14,15);
}

// B-operand fragment from transposed LDS buffer: buf[n][k], stride LDP
__device__ __forceinline__ v16h load_Bfrag(const _Float16* buf, int lane, int kb, int n0) {
    int row = n0 + (lane & 15);
    int kh  = kb + ((lane >> 4) << 3);
    const _Float16* p = buf + row * LDP + kh;
    v8h lo = *(const v8h*)(p);
    v8h hi = *(const v8h*)(p + 16);
    return __builtin_shufflevector(lo, hi, 0,1,2,3,4,5,6,7,8,9,10,11,12,13,14,15);
}

// store 16x16 f32 accumulator tile transposed into LDS as f16: element (m,n) -> buf[n*LDP+m]
__device__ __forceinline__ void store_acc_tr(_Float16* buf, v8f c, int lane, int m0, int n0,
                                             float sc) {
    int n  = n0 + (lane & 15);
    int mo = m0 + ((lane >> 4) << 3);     // C/D layout: vgpr r -> M = r + 8*half
    v8h h;
#pragma unroll
    for (int r = 0; r < 8; ++r) h[r] = (_Float16)(c[r] * sc);
    *(v8h*)(buf + n * LDP + mo) = h;      // 16B contiguous per lane
}

__global__ __launch_bounds__(512) void wht_conv2d_kernel(
    const float* __restrict__ x, const float* __restrict__ v,
    const float* __restrict__ T, float* __restrict__ out) {
    extern __shared__ _Float16 smem[];
    _Float16* bufR = smem;               // row-major buffer (X, then F6)
    _Float16* bufT = smem + 256 * LDP;   // transposed buffer (Y^T, then G^T)

    const int  tid  = threadIdx.x;
    const int  wave = tid >> 5;
    const int  lane = tid & 31;
    const long ioff = (long)blockIdx.x * IMG;
    const float sc  = 0.0625f;           // 1/sqrt(256) per 1-D transform

    // ---- load X (fp32 global) -> f16 row-major LDS ----
    const float4* xin = (const float4*)(x + ioff);
    for (int i = tid; i < IMG / 4; i += 512) {
        float4 q = xin[i];
        int e = i << 2;
        v4h h; h[0] = (_Float16)q.x; h[1] = (_Float16)q.y;
               h[2] = (_Float16)q.z; h[3] = (_Float16)q.w;
        *(v4h*)(bufR + (e >> 8) * LDP + (e & 255)) = h;
    }
    __syncthreads();

    // ---- Stage A: Y = (X * H) * sc ; store Y^T ----
    {
        int n0 = wave << 4;
        v16h hB[8];
#pragma unroll
        for (int ks = 0; ks < 8; ++ks) hB[ks] = make_hfrag_B(lane, ks << 5, n0);
        for (int mt = 0; mt < 16; ++mt) {
            int m0 = mt << 4;
            v8f acc = {0.f, 0.f, 0.f, 0.f, 0.f, 0.f, 0.f, 0.f};
#pragma unroll
            for (int ks = 0; ks < 8; ++ks) {
                v16h a = load_Afrag(bufR, lane, m0, ks << 5);
                acc = __builtin_amdgcn_wmma_f32_16x16x32_f16(false, a, false, hB[ks],
                                                             (short)0, acc, false, false);
            }
            store_acc_tr(bufT, acc, lane, m0, n0, sc);
        }
    }
    __syncthreads();

    // ---- Stage B: F2 = (H * Y) * sc, fused pointwise -> F6 row-major ----
    {
        int m0 = wave << 4;
        v16h hA[8];
#pragma unroll
        for (int ks = 0; ks < 8; ++ks) hA[ks] = make_hfrag_A(lane, m0, ks << 5);
        int wl = lane & 15;
        int half8 = (lane >> 4) << 3;
        for (int nt = 0; nt < 16; ++nt) {
            int n0 = nt << 4;
            v8f acc = {0.f, 0.f, 0.f, 0.f, 0.f, 0.f, 0.f, 0.f};
#pragma unroll
            for (int ks = 0; ks < 8; ++ks) {
                v16h b = load_Bfrag(bufT, lane, ks << 5, n0);
                acc = __builtin_amdgcn_wmma_f32_16x16x32_f16(false, hA[ks], false, b,
                                                             (short)0, acc, false, false);
            }
            int w = n0 + wl;
#pragma unroll
            for (int r = 0; r < 8; ++r) {
                int h = m0 + half8 + r;
                float f2 = acc[r] * sc;
                float s = 0.f;
#pragma unroll
                for (int p = 0; p < 4; ++p) {
                    int idx = (p << 16) | (h << 8) | w;   // v,T are (4,256,256), L2-resident
                    float a = f2 * v[idx];
                    float g = fabsf(a) - fabsf(T[idx]);
                    s += tanhf(a) * fmaxf(g, 0.f);
                }
                bufR[h * LDP + w] = (_Float16)s;
            }
        }
    }
    __syncthreads();

    // ---- Stage C: G = (F6 * H) * sc ; store G^T ----
    {
        int n0 = wave << 4;
        v16h hB[8];
#pragma unroll
        for (int ks = 0; ks < 8; ++ks) hB[ks] = make_hfrag_B(lane, ks << 5, n0);
        for (int mt = 0; mt < 16; ++mt) {
            int m0 = mt << 4;
            v8f acc = {0.f, 0.f, 0.f, 0.f, 0.f, 0.f, 0.f, 0.f};
#pragma unroll
            for (int ks = 0; ks < 8; ++ks) {
                v16h a = load_Afrag(bufR, lane, m0, ks << 5);
                acc = __builtin_amdgcn_wmma_f32_16x16x32_f16(false, a, false, hB[ks],
                                                             (short)0, acc, false, false);
            }
            store_acc_tr(bufT, acc, lane, m0, n0, sc);
        }
    }
    __syncthreads();

    // ---- Stage D: OUT = (H * G) * sc + X ----
    {
        int m0 = wave << 4;
        v16h hA[8];
#pragma unroll
        for (int ks = 0; ks < 8; ++ks) hA[ks] = make_hfrag_A(lane, m0, ks << 5);
        int wl = lane & 15;
        int half8 = (lane >> 4) << 3;
        for (int nt = 0; nt < 16; ++nt) {
            int n0 = nt << 4;
            v8f acc = {0.f, 0.f, 0.f, 0.f, 0.f, 0.f, 0.f, 0.f};
#pragma unroll
            for (int ks = 0; ks < 8; ++ks) {
                v16h b = load_Bfrag(bufT, lane, ks << 5, n0);
                acc = __builtin_amdgcn_wmma_f32_16x16x32_f16(false, hA[ks], false, b,
                                                             (short)0, acc, false, false);
            }
            int w = n0 + wl;
#pragma unroll
            for (int r = 0; r < 8; ++r) {
                int  h   = m0 + half8 + r;
                long idx = ioff + (long)((h << 8) | w);
                out[idx] = acc[r] * sc + x[idx];   // residual in fp32
            }
        }
    }
}

extern "C" void kernel_launch(void* const* d_in, const int* in_sizes, int n_in,
                              void* d_out, int out_size, void* d_ws, size_t ws_size,
                              hipStream_t stream) {
    const float* x = (const float*)d_in[0];
    const float* v = (const float*)d_in[1];
    const float* T = (const float*)d_in[2];
    float* out = (float*)d_out;

    const size_t shmem = (size_t)2 * 256 * LDP * sizeof(_Float16);  // 264 KB (<= 320 KB WGP LDS)
    static bool attr_set_dummy = []() {
        // opt-in to large dynamic LDS; deterministic, no stream interaction
        (void)hipFuncSetAttribute((const void*)wht_conv2d_kernel,
                                  hipFuncAttributeMaxDynamicSharedMemorySize,
                                  (int)((size_t)2 * 256 * LDP * sizeof(_Float16)));
        return true;
    }();
    (void)attr_set_dummy;

    wht_conv2d_kernel<<<512, 512, shmem, stream>>>(x, v, T, out);
}
